// SlotAttentionGroupingIA3_58076547776891
// MI455X (gfx1250) — matrix operand order repaired and checked by reading, not compiled, not run on hardware
//
#include <hip/hip_runtime.h>
#include <hip/hip_bf16.h>

#define DEV __device__ __forceinline__

typedef __bf16 bf16;
typedef bf16  v16bf __attribute__((ext_vector_type(16)));
typedef float v8f   __attribute__((ext_vector_type(8)));

// ---------------- constants ----------------
// B=8, N=4096, F=768, K(slots)=16, D=256, HID=1024, ITERS=3
#define BB 8
#define NN 4096
#define FF 768
#define KS 16
#define DD 256
#define HID 1024
#define EPSV 1e-8f
#define SCALE 0.0625f   // 256^-0.5

// ---------------- WMMA helpers (wave32, 16x16x32 bf16) ----------------
DEV v8f zero8() {
    v8f z;
#pragma unroll
    for (int j = 0; j < 8; ++j) z[j] = 0.f;
    return z;
}

DEV v8f wmma_bf16(v16bf a, v16bf b, v8f c) {
    return __builtin_amdgcn_wmma_f32_16x16x32_bf16(
        /*neg_a=*/false, a, /*neg_b=*/false, b,
        /*c_mod=*/(short)0, c, /*reuse_a=*/false, /*reuse_b=*/false);
}

union frag_cast { uint4 u[2]; v16bf v; };

// A fragment: 16x32 (MxK) tile, row-major with stride ld (elements).
// Lane l: m=l&15, kh=l>>4. Elements 0..7 <-> K=8kh+e ; 8..15 <-> K=16+8kh+(e-8).
// Both runs are 16B-contiguous -> two 128-bit loads.
DEV v16bf load_a_frag(const bf16* src, int ld) {
    int lane = threadIdx.x & 31;
    const bf16* p = src + (size_t)(lane & 15) * ld + 8 * (lane >> 4);
    frag_cast cv;
    cv.u[0] = *(const uint4*)p;
    cv.u[1] = *(const uint4*)(p + 16);
    return cv.v;
}

// B fragment: 32x16 (KxN) tile stored COLUMN-major (n-major): src[n*ld + k].
// Lane l: n=l&15, kh=l>>4. Elements 0..7 <-> K=8kh+e ; 8..15 <-> K=16+8kh+(e-8).
DEV v16bf load_b_frag_nmajor(const bf16* src, int ld) {
    int lane = threadIdx.x & 31;
    const bf16* p = src + (size_t)(lane & 15) * ld + 8 * (lane >> 4);
    frag_cast cv;
    cv.u[0] = *(const uint4*)p;
    cv.u[1] = *(const uint4*)(p + 16);
    return cv.v;
}

// block-wide sum over 256 threads
DEV float block_sum(float v, float* red) {
    int t = threadIdx.x;
    red[t] = v;
    __syncthreads();
    for (int off = 128; off > 0; off >>= 1) {
        if (t < off) red[t] += red[t + off];
        __syncthreads();
    }
    float s = red[0];
    __syncthreads();
    return s;
}

// ---------------- kernel 0: Wk|Wv fp32 -> bf16 TRANSPOSED [512][768] ----------------
__global__ __launch_bounds__(256) void prep_wkv_kernel(const float* __restrict__ Wk,
                                                       const float* __restrict__ Wv,
                                                       bf16* __restrict__ wkvT) {
    int c = blockIdx.x;        // output col 0..511
    int t = threadIdx.x;
#pragma unroll
    for (int rep = 0; rep < 3; ++rep) {
        int r = rep * 256 + t;  // 0..767
        float val = (c < DD) ? Wk[(size_t)r * DD + c] : Wv[(size_t)r * DD + (c - DD)];
        wkvT[(size_t)c * FF + r] = (bf16)val;
    }
}

// ---------------- kernel 1: LN(feature) -> bf16 xln [32768][768] ----------------
__global__ __launch_bounds__(256) void ln_in_kernel(const float* __restrict__ feat,
                                                    const float* __restrict__ g,
                                                    const float* __restrict__ bta,
                                                    bf16* __restrict__ xln) {
    __shared__ float red[256];
    int row = blockIdx.x;
    int t = threadIdx.x;
    const float* src = feat + (size_t)row * FF;
    float v0 = src[t], v1 = src[t + 256], v2 = src[t + 512];
    float mu = block_sum(v0 + v1 + v2, red) * (1.f / FF);
    float d0 = v0 - mu, d1 = v1 - mu, d2 = v2 - mu;
    float var = block_sum(d0 * d0 + d1 * d1 + d2 * d2, red) * (1.f / FF);
    float rstd = rsqrtf(var + 1e-5f);
    bf16* dst = xln + (size_t)row * FF;
    dst[t]       = (bf16)(g[t] * d0 * rstd + bta[t]);
    dst[t + 256] = (bf16)(g[t + 256] * d1 * rstd + bta[t + 256]);
    dst[t + 512] = (bf16)(g[t + 512] * d2 * rstd + bta[t + 512]);
}

// ---------------- kernel 2: IA3 scales + initial slots ----------------
__global__ __launch_bounds__(256) void scales_init_kernel(
    const float* __restrict__ code, const float* __restrict__ cond,
    const float* __restrict__ Wqs, const float* __restrict__ bqs,
    const float* __restrict__ Wvs, const float* __restrict__ bvs,
    const float* __restrict__ Wfs, const float* __restrict__ bfs,
    float* __restrict__ qs, float* __restrict__ vs, float* __restrict__ fs,
    float* __restrict__ slots) {
    __shared__ float c[DD];
    int rk = blockIdx.x;  // 0..127 = b*16+k
    int t = threadIdx.x;
    c[t] = code[(size_t)rk * DD + t];
    slots[(size_t)rk * DD + t] = cond[(size_t)rk * DD + t];
    __syncthreads();
    float aq = bqs[t], av = bvs[t];
    for (int j = 0; j < DD; ++j) {
        float cj = c[j];
        aq += cj * Wqs[(size_t)j * DD + t];
        av += cj * Wvs[(size_t)j * DD + t];
    }
    qs[(size_t)rk * DD + t] = aq;
    vs[(size_t)rk * DD + t] = av;
#pragma unroll
    for (int rep = 0; rep < 4; ++rep) {
        int o = rep * 256 + t;
        float a = bfs[o];
        for (int j = 0; j < DD; ++j) a += c[j] * Wfs[(size_t)j * HID + o];
        fs[(size_t)rk * HID + o] = a;
    }
}

// ---------------- kernel 3: kv GEMM: [32768x768]bf16 @ [768x512]bf16 ----------------
// cols 0..255 -> kmat[b][n][d] bf16 ; cols 256..511 -> vT[b][d][n] bf16
__global__ __launch_bounds__(256) void kv_gemm_kernel(const bf16* __restrict__ xln,
                                                      const bf16* __restrict__ wkvT,
                                                      bf16* __restrict__ kmat,
                                                      bf16* __restrict__ vT) {
    __shared__ __align__(16) bf16 As[128 * 32];   // row-major [m][k]
    __shared__ __align__(16) bf16 Bs[64 * 32];    // n-major   [n][k]
    int wgM = blockIdx.x >> 3;   // 256 M-tiles of 128
    int wgN = blockIdx.x & 7;    // 8 N-tiles of 64
    int tid = threadIdx.x;
    int wave = tid >> 5, lane = tid & 31;
    int mw = wave & 3, nw = wave >> 2;   // 4x2 wave grid, each wave 32x32
    int m0g = wgM * 128, n0g = wgN * 64;

    v8f acc00 = zero8(), acc01 = zero8(), acc10 = zero8(), acc11 = zero8();

    for (int k0 = 0; k0 < FF; k0 += 32) {
        {   // A tile 128x32: each thread copies 32B
            int r = tid >> 1, cc = (tid & 1) * 16;
            const uint4* s = (const uint4*)&xln[(size_t)(m0g + r) * FF + k0 + cc];
            uint4* d = (uint4*)&As[r * 32 + cc];
            d[0] = s[0]; d[1] = s[1];
            if (k0 + 32 < FF)   // software prefetch of next chunk
                __builtin_prefetch(&xln[(size_t)(m0g + r) * FF + k0 + 32 + cc], 0, 1);
        }
        {   // B tile (n-major) 64x32: each thread copies 16B
            int r = tid >> 2, cc = (tid & 3) * 8;
            *(uint4*)&Bs[r * 32 + cc] =
                *(const uint4*)&wkvT[(size_t)(n0g + r) * FF + k0 + cc];
        }
        __syncthreads();
        v16bf a0 = load_a_frag(&As[(mw * 32 + 0) * 32], 32);
        v16bf a1 = load_a_frag(&As[(mw * 32 + 16) * 32], 32);
        v16bf b0 = load_b_frag_nmajor(&Bs[(nw * 32 + 0) * 32], 32);
        v16bf b1 = load_b_frag_nmajor(&Bs[(nw * 32 + 16) * 32], 32);
        acc00 = wmma_bf16(a0, b0, acc00);
        acc01 = wmma_bf16(a0, b1, acc01);
        acc10 = wmma_bf16(a1, b0, acc10);
        acc11 = wmma_bf16(a1, b1, acc11);
        __syncthreads();
    }

    int nloc = lane & 15, hi = lane >> 4;
    v8f* accs[4] = {&acc00, &acc01, &acc10, &acc11};
#pragma unroll
    for (int q = 0; q < 4; ++q) {
        int tm = q >> 1, tn = q & 1;
        int colbase = n0g + nw * 32 + tn * 16;   // wave-uniform -> scalar branch
        if (colbase < DD) {
#pragma unroll
            for (int j = 0; j < 8; ++j) {
                int gm = m0g + mw * 32 + tm * 16 + j + 8 * hi;
                int b = gm >> 12, n = gm & 4095;
                kmat[((size_t)b * NN + n) * DD + colbase + nloc] = (bf16)(*accs[q])[j];
            }
        } else {
#pragma unroll
            for (int j = 0; j < 8; ++j) {
                int gm = m0g + mw * 32 + tm * 16 + j + 8 * hi;
                int b = gm >> 12, n = gm & 4095;
                vT[((size_t)b * DD + colbase - DD + nloc) * NN + n] = (bf16)(*accs[q])[j];
            }
        }
    }
}

// ---------------- kernel 4 (per iter): LN(slots), q=(s@Wq)*q_scale -> bf16 ----------------
__global__ __launch_bounds__(256) void slot_prep_kernel(const float* __restrict__ slots,
                                                        const float* __restrict__ g,
                                                        const float* __restrict__ bt,
                                                        const float* __restrict__ Wq,
                                                        const float* __restrict__ qs,
                                                        bf16* __restrict__ qbf,
                                                        float* __restrict__ rowsum) {
    __shared__ float s2[DD];
    __shared__ float red[256];
    int rk = blockIdx.x, t = threadIdx.x;
    float v = slots[(size_t)rk * DD + t];
    float mu = block_sum(v, red) * (1.f / DD);
    float d = v - mu;
    float var = block_sum(d * d, red) * (1.f / DD);
    float rstd = rsqrtf(var + 1e-5f);
    s2[t] = g[t] * d * rstd + bt[t];
    if (t == 0) rowsum[rk] = 0.f;
    __syncthreads();
    float acc = 0.f;
    for (int j = 0; j < DD; ++j) acc += s2[j] * Wq[(size_t)j * DD + t];
    qbf[(size_t)rk * DD + t] = (bf16)(acc * qs[(size_t)rk * DD + t]);
}

// ---------------- kernel 5 (per iter): dots = q@k^T, softmax over slots, rowsums ----
// B operand: kmat[b][n][d] (n-major w.r.t. the K=d dimension) -> contiguous frags.
__global__ __launch_bounds__(256) void attn_dots_kernel(const bf16* __restrict__ qbf,
                                                        const bf16* __restrict__ kmat,
                                                        float* __restrict__ attn_out,
                                                        bf16* __restrict__ attnb,
                                                        float* __restrict__ rowsum) {
    int b = blockIdx.x >> 5;      // 8 batches
    int chunk = blockIdx.x & 31;  // 32 chunks x 128 cols
    int wave = threadIdx.x >> 5, lane = threadIdx.x & 31;
    int n0 = chunk * 128 + wave * 16;
    const bf16* q  = qbf + (size_t)b * KS * DD;
    const bf16* km = kmat + (size_t)b * NN * DD;

    v8f acc = zero8();
#pragma unroll
    for (int k0 = 0; k0 < DD; k0 += 32) {
        v16bf a  = load_a_frag(q + k0, DD);
        v16bf bb = load_b_frag_nmajor(km + (size_t)n0 * DD + k0, DD);
        acc = wmma_bf16(a, bb, acc);
    }
#pragma unroll
    for (int j = 0; j < 8; ++j) acc[j] *= SCALE;

    // softmax across the 16 slots of this column: lanes l and l^16 hold the two row-halves
    float mx = acc[0];
#pragma unroll
    for (int j = 1; j < 8; ++j) mx = fmaxf(mx, acc[j]);
    mx = fmaxf(mx, __shfl_xor(mx, 16));
    float e[8], s = 0.f;
#pragma unroll
    for (int j = 0; j < 8; ++j) { e[j] = __expf(acc[j] - mx); s += e[j]; }
    s += __shfl_xor(s, 16);
    float inv = 1.f / s;

    int n = n0 + (lane & 15), hi = lane >> 4;
#pragma unroll
    for (int j = 0; j < 8; ++j) {
        int m = j + 8 * hi;
        float p = e[j] * inv;
        attn_out[((size_t)b * KS + m) * NN + n] = p;       // attn_before
        float pe = p + EPSV;
        attnb[((size_t)b * KS + m) * NN + n] = (bf16)pe;
        float v = pe;                                       // row-sum over 16 cols of half
        v += __shfl_xor(v, 1);
        v += __shfl_xor(v, 2);
        v += __shfl_xor(v, 4);
        v += __shfl_xor(v, 8);
        if ((lane & 15) == 0) atomicAdd(&rowsum[b * KS + m], v);
    }
}

// ---------------- kernel 6 (per iter): updates = vscale * (attn/rowsum) @ v --------
// B operand: vT[b][d][n] -> per-lane contiguous K(=n) runs.
__global__ __launch_bounds__(256) void updates_gemm_kernel(const bf16* __restrict__ attnb,
                                                           const bf16* __restrict__ vT,
                                                           const float* __restrict__ rowsum,
                                                           const float* __restrict__ vs,
                                                           float* __restrict__ upd) {
    int b = blockIdx.x;
    int wave = threadIdx.x >> 5, lane = threadIdx.x & 31;
    const bf16* A = attnb + (size_t)b * KS * NN;
    const bf16* V = vT + (size_t)b * DD * NN;
#pragma unroll
    for (int t = 0; t < 2; ++t) {
        int d0 = (wave + t * 8) * 16;    // d-tile
        v8f acc = zero8();
        for (int k0 = 0; k0 < NN; k0 += 32) {
            v16bf a  = load_a_frag(A + k0, NN);
            v16bf bb = load_b_frag_nmajor(V + (size_t)d0 * NN + k0, NN);
            acc = wmma_bf16(a, bb, acc);
        }
        int d = d0 + (lane & 15), hi = lane >> 4;
#pragma unroll
        for (int j = 0; j < 8; ++j) {
            int m = j + 8 * hi;
            float val = acc[j] / rowsum[b * KS + m] * vs[((size_t)b * KS + m) * DD + d];
            upd[((size_t)b * KS + m) * DD + d] = val;
        }
    }
}

// ---------------- kernel 7 (per iter): GRU cell + gated FF (per slot row) ----------------
__global__ __launch_bounds__(256) void gru_ff_kernel(
    const float* __restrict__ upd, float* __restrict__ slots,
    const float* __restrict__ W_ih, const float* __restrict__ b_ih,
    const float* __restrict__ W_hh, const float* __restrict__ b_hh,
    const float* __restrict__ lng, const float* __restrict__ lnb,
    const float* __restrict__ W1, const float* __restrict__ b1,
    const float* __restrict__ W2, const float* __restrict__ b2,
    const float* __restrict__ fs, float* __restrict__ out_slots) {
    __shared__ float u[DD], hp[DD], xg[768], hg[768], s2[DD], h1[HID];
    __shared__ float red[256];
    int rk = blockIdx.x, t = threadIdx.x;
    u[t]  = upd[(size_t)rk * DD + t];
    hp[t] = slots[(size_t)rk * DD + t];
    __syncthreads();
#pragma unroll
    for (int rep = 0; rep < 3; ++rep) {
        int o = rep * 256 + t;
        float ax = b_ih[o], ah = b_hh[o];
        for (int j = 0; j < DD; ++j) {
            ax += u[j]  * W_ih[(size_t)j * 768 + o];
            ah += hp[j] * W_hh[(size_t)j * 768 + o];
        }
        xg[o] = ax;
        hg[o] = ah;
    }
    __syncthreads();
    float r  = 1.f / (1.f + __expf(-(xg[t] + hg[t])));
    float z  = 1.f / (1.f + __expf(-(xg[256 + t] + hg[256 + t])));
    float nn = tanhf(xg[512 + t] + r * hg[512 + t]);
    float h  = (1.f - z) * nn + z * hp[t];
    // FF layer-norm
    float mu = block_sum(h, red) * (1.f / DD);
    float d = h - mu;
    float var = block_sum(d * d, red) * (1.f / DD);
    float rstd = rsqrtf(var + 1e-5f);
    s2[t] = lng[t] * d * rstd + lnb[t];
    __syncthreads();
#pragma unroll
    for (int rep = 0; rep < 4; ++rep) {
        int o = rep * 256 + t;
        float a = b1[o];
        for (int j = 0; j < DD; ++j) a += s2[j] * W1[(size_t)j * HID + o];
        h1[o] = fmaxf(a, 0.f) * fs[(size_t)rk * HID + o];
    }
    __syncthreads();
    float a = b2[t];
    for (int o = 0; o < HID; ++o) a += h1[o] * W2[(size_t)o * DD + t];
    float res = a + h;
    slots[(size_t)rk * DD + t] = res;
    if (out_slots) out_slots[(size_t)rk * DD + t] = res;
}

// ---------------- host ----------------
extern "C" void kernel_launch(void* const* d_in, const int* in_sizes, int n_in,
                              void* d_out, int out_size, void* d_ws, size_t ws_size,
                              hipStream_t stream) {
    // inputs (setup_inputs dict order, params flattened in key order)
    const float* feature = (const float*)d_in[0];
    const float* cond    = (const float*)d_in[1];
    const float* code    = (const float*)d_in[2];
    const float* ln_in_g = (const float*)d_in[3];
    const float* ln_in_b = (const float*)d_in[4];
    const float* ln_s_g  = (const float*)d_in[5];
    const float* ln_s_b  = (const float*)d_in[6];
    const float* ln_ff_g = (const float*)d_in[7];
    const float* ln_ff_b = (const float*)d_in[8];
    const float* Wq      = (const float*)d_in[9];
    const float* Wk      = (const float*)d_in[10];
    const float* Wv      = (const float*)d_in[11];
    const float* W_ih    = (const float*)d_in[12];
    const float* b_ih    = (const float*)d_in[13];
    const float* W_hh    = (const float*)d_in[14];
    const float* b_hh    = (const float*)d_in[15];
    const float* W1      = (const float*)d_in[16];
    const float* b1      = (const float*)d_in[17];
    const float* W2      = (const float*)d_in[18];
    const float* b2      = (const float*)d_in[19];
    const float* Wqs     = (const float*)d_in[20];
    const float* bqs     = (const float*)d_in[21];
    const float* Wvs     = (const float*)d_in[22];
    const float* bvs     = (const float*)d_in[23];
    const float* Wfs     = (const float*)d_in[24];
    const float* bfs     = (const float*)d_in[25];
    (void)in_sizes; (void)n_in; (void)out_size; (void)ws_size;

    float* out = (float*)d_out;
    float* out_slots = out;                // [8,16,256]
    float* out_attn  = out + BB * KS * DD; // [8,16,4096]

    // workspace carve-up
    char* w = (char*)d_ws;
    size_t off = 0;
    auto take = [&](size_t bytes) {
        void* p = (void*)(w + off);
        off += (bytes + 255) & ~(size_t)255;
        return p;
    };
    bf16*  xln    = (bf16*) take((size_t)BB * NN * FF * 2);
    bf16*  wkvT   = (bf16*) take((size_t)512 * FF * 2);
    bf16*  kmat   = (bf16*) take((size_t)BB * NN * DD * 2);
    bf16*  vT     = (bf16*) take((size_t)BB * DD * NN * 2);
    float* qs     = (float*)take((size_t)BB * KS * DD * 4);
    float* vs     = (float*)take((size_t)BB * KS * DD * 4);
    float* fsbuf  = (float*)take((size_t)BB * KS * HID * 4);
    float* slots  = (float*)take((size_t)BB * KS * DD * 4);
    bf16*  qbf    = (bf16*) take((size_t)BB * KS * DD * 2);
    bf16*  attnb  = (bf16*) take((size_t)BB * KS * NN * 2);
    float* rowsum = (float*)take((size_t)BB * KS * 4);
    float* upd    = (float*)take((size_t)BB * KS * DD * 4);

    // one-time prep
    prep_wkv_kernel<<<512, 256, 0, stream>>>(Wk, Wv, wkvT);
    ln_in_kernel<<<BB * NN, 256, 0, stream>>>(feature, ln_in_g, ln_in_b, xln);
    scales_init_kernel<<<BB * KS, 256, 0, stream>>>(code, cond, Wqs, bqs, Wvs, bvs,
                                                    Wfs, bfs, qs, vs, fsbuf, slots);
    kv_gemm_kernel<<<(BB * NN / 128) * (512 / 64), 256, 0, stream>>>(xln, wkvT, kmat, vT);

    // 3 slot-attention iterations
    for (int it = 0; it < 3; ++it) {
        slot_prep_kernel<<<BB * KS, 256, 0, stream>>>(slots, ln_s_g, ln_s_b, Wq, qs,
                                                      qbf, rowsum);
        attn_dots_kernel<<<BB * 32, 256, 0, stream>>>(qbf, kmat, out_attn, attnb, rowsum);
        updates_gemm_kernel<<<BB, 256, 0, stream>>>(attnb, vT, rowsum, vs, upd);
        gru_ff_kernel<<<BB * KS, 256, 0, stream>>>(upd, slots, W_ih, b_ih, W_hh, b_hh,
                                                   ln_ff_g, ln_ff_b, W1, b1, W2, b2,
                                                   fsbuf, (it == 2) ? out_slots : nullptr);
    }
}
